// VAELSTM_90056874263186
// MI455X (gfx1250) — compile-verified, matrix operand order checked by reading, not run on hardware
//
#include <hip/hip_runtime.h>
#include <hip/hip_bf16.h>

// VAE-LSTM forward for MI455X (gfx1250), wave32 + WMMA f16 (16x16x32, f32 acc).
// One workgroup (4 waves / 128 threads) owns 16 batch rows persistently:
// encoder LSTM (30 steps) -> mean/logvar MLPs -> latent -> decoder LSTM (30 steps).
// v2: branchless exp2/rcp transcendentals, hoisted B-fragments (weights held in
// VGPRs across the time loops), ping-pong LDS h/emb buffers (1 barrier/step in
// the encoder), W_se/b_se/W_c/b_c cached in LDS.

#define TSTEPS 30
#define HID    64
#define EMB    64
#define LAT    100
#define LATP   112   // LAT padded to N multiple of 16
#define LATK   128   // LAT padded to K multiple of 32
#define DTC    0.03f
#define LOG2E  1.4426950408889634f

typedef __attribute__((ext_vector_type(16))) _Float16 v16h;
typedef __attribute__((ext_vector_type(8)))  _Float16 v8h;
typedef __attribute__((ext_vector_type(8)))  float    v8f;

static __device__ __forceinline__ float fexp_(float x)  { return __builtin_amdgcn_exp2f(x * LOG2E); }
static __device__ __forceinline__ float frcp_(float x)  { return __builtin_amdgcn_rcpf(x); }
// sigmoid(x) = 1/(1+e^-x); branchless, saturates via exp2->inf/0.
static __device__ __forceinline__ float sigmoidf_(float x) { return frcp_(1.0f + fexp_(-x)); }
// tanh(x) = 1 - 2/(e^{2x}+1); x->+inf: exp2->inf, rcp->0 => 1; x->-inf => -1.
static __device__ __forceinline__ float tanh_(float x) { return 1.0f - 2.0f * frcp_(1.0f + fexp_(2.0f * x)); }
static __device__ __forceinline__ float leakyf_(float x) { return x >= 0.0f ? x : 0.01f * x; }

static __device__ __forceinline__ v8f splat8(float x) {
    v8f v;
#pragma unroll
    for (int i = 0; i < 8; ++i) v[i] = x;
    return v;
}

// Build a 16-half fragment from two contiguous 16B chunks (ISA 16-bit layout:
// lanes 0-15 hold K kb+[0,8) and K 16+kb+[0,8) with kb=0; lanes 16-31 kb=8).
static __device__ __forceinline__ v16h frag_ld(const _Float16* p) {
    union { v16h v; v8h h[2]; } u;
    u.h[0] = *(const v8h*)(p);
    u.h[1] = *(const v8h*)(p + 16);
    return u.v;
}

// A fragment: activations, row-major f16 [16][stride], K-tile kt.
static __device__ __forceinline__ v16h frag_a(const _Float16* base, int stride, int lane, int kt) {
    int m  = lane & 15;
    int kb = (lane >> 4) * 8;
    return frag_ld(base + m * stride + kt * 32 + kb);
}

// B fragment: B = W^T with W row-major f16 [N][stride]; tile (nt, kt).
// Lane l holds column n = nt*16 + (l&15) of B = row n of W, contiguous in K.
static __device__ __forceinline__ v16h frag_b(const _Float16* base, int stride, int lane, int nt, int kt) {
    int n  = nt * 16 + (lane & 15);
    int kb = (lane >> 4) * 8;
    return frag_ld(base + n * stride + kt * 32 + kb);
}

static __device__ __forceinline__ v8f wmma16(v16h a, v16h b, v8f c) {
    // (neg_a, A, neg_b, B, c_mod, C, reuse_a, reuse_b)
    return __builtin_amdgcn_wmma_f32_16x16x32_f16(false, a, false, b, (short)0, c, false, false);
}

// --- weight conversion: f32 [N][K] -> f16 [Npad][Kpad], zero padded -------
__global__ void cvt_pad_kernel(const float* __restrict__ src, _Float16* __restrict__ dst,
                               int N, int K, int Npad, int Kpad) {
    int idx = blockIdx.x * blockDim.x + threadIdx.x;
    int total = Npad * Kpad;
    if (idx >= total) return;
    int n = idx / Kpad, k = idx % Kpad;
    float v = (n < N && k < K) ? src[n * K + k] : 0.0f;
    dst[idx] = (_Float16)v;
}

// --------------------------- main persistent kernel -----------------------
__global__ __launch_bounds__(128)
void vae_lstm_kernel(const float* __restrict__ traj, const float* __restrict__ init_state,
                     const float* __restrict__ eps,
                     const float* __restrict__ W_se, const float* __restrict__ b_se,
                     const float* __restrict__ b_ih_e, const float* __restrict__ b_hh_e,
                     const float* __restrict__ mean_bs, const float* __restrict__ mean_bf,
                     const float* __restrict__ lv_bs, const float* __restrict__ lv_bf,
                     const float* __restrict__ b_init,
                     const float* __restrict__ b_ih_d, const float* __restrict__ b_hh_d,
                     const float* __restrict__ W_c, const float* __restrict__ b_c,
                     const _Float16* __restrict__ wIhE, const _Float16* __restrict__ wHhE,
                     const _Float16* __restrict__ wMeanS, const _Float16* __restrict__ wMeanF,
                     const _Float16* __restrict__ wLvS, const _Float16* __restrict__ wLvF,
                     const _Float16* __restrict__ wInit,
                     const _Float16* __restrict__ wIhD, const _Float16* __restrict__ wHhD,
                     float* __restrict__ out, int B) {
    const int b0   = blockIdx.x * 16;
    const int tid  = threadIdx.x;
    const int wave = tid >> 5;
    const int lane = tid & 31;

    float* out_rec = out;
    float* out_exp = out + (size_t)B * TSTEPS * 4;
    float* out_mu  = out + (size_t)B * TSTEPS * 4 * 2;
    float* out_lv  = out_mu + (size_t)B * LAT;

    __shared__ __align__(16) float    traj_s[16 * TSTEPS * 4];
    __shared__ __align__(16) _Float16 hping[2][16 * HID];
    __shared__ __align__(16) _Float16 embp[2][16 * EMB];
    __shared__ __align__(16) _Float16 m0[16 * HID];
    __shared__ __align__(16) _Float16 m1[16 * HID];
    __shared__ __align__(16) float    mu_s[16 * LATP];
    __shared__ __align__(16) float    lv_s[16 * LATP];
    __shared__ __align__(16) _Float16 zbuf[16 * LATK];
    __shared__ __align__(16) float    prev_s[16 * 4];
    __shared__ __align__(16) float    wse_s[EMB * 4];
    __shared__ __align__(16) float    bse_s[EMB];
    __shared__ __align__(16) float    wc_s[2 * HID];
    __shared__ __align__(16) float    bc_s[2];

    // Load trajectory tile; pass through expert_traj output; cache small weights.
    for (int i = tid; i < 16 * TSTEPS * 4; i += 128) {
        int r = i / (TSTEPS * 4), c = i % (TSTEPS * 4);
        float v = traj[(size_t)(b0 + r) * (TSTEPS * 4) + c];
        traj_s[i] = v;
        out_exp[(size_t)(b0 + r) * (TSTEPS * 4) + c] = v;
    }
    for (int i = tid; i < 16 * HID; i += 128) hping[0][i] = (_Float16)0.0f;
    for (int i = tid; i < EMB * 4; i += 128) wse_s[i] = W_se[i];
    for (int i = tid; i < 2 * HID; i += 128) wc_s[i] = W_c[i];
    if (tid < EMB) bse_s[tid] = b_se[tid];
    if (tid < 2)   bc_s[tid]  = b_c[tid];
    __syncthreads();

    float creg[8];
#pragma unroll
    for (int r = 0; r < 8; ++r) creg[r] = 0.0f;
    const int kb  = (lane >> 4) * 8;
    const int col = wave * 16 + (lane & 15);

    // ----------------------------- encoder -------------------------------
    {
        // Hoist B fragments (weights) and gate biases into registers.
        v16h bIh[4][2], bHh[4][2];
        float gbias[4];
#pragma unroll
        for (int g = 0; g < 4; ++g) {
            int nt = g * 4 + wave;
            int nl = nt * 16 + (lane & 15);
            gbias[g] = b_ih_e[nl] + b_hh_e[nl];
#pragma unroll
            for (int kt = 0; kt < 2; ++kt) {
                bIh[g][kt] = frag_b(wIhE, 64, lane, nt, kt);
                bHh[g][kt] = frag_b(wHhE, 64, lane, nt, kt);
            }
        }
        for (int t = 0; t < TSTEPS; ++t) {
            _Float16* eb = embp[t & 1];
            for (int u = tid; u < 16 * EMB; u += 128) {   // emb = inp @ W_se.T + b_se (K=4)
                int r = u >> 6, c = u & 63;
                const float* tr = traj_s + r * 120;
                float i0, i1;
                if (t == 0) { i0 = tr[0]; i1 = tr[1]; }
                else        { i0 = tr[t * 4] - tr[(t - 1) * 4]; i1 = tr[t * 4 + 1] - tr[(t - 1) * 4 + 1]; }
                float i2 = tr[t * 4 + 2], i3 = tr[t * 4 + 3];
                const float* wr = wse_s + c * 4;
                eb[u] = (_Float16)(bse_s[c] + i0 * wr[0] + i1 * wr[1] + i2 * wr[2] + i3 * wr[3]);
            }
            __syncthreads();   // single barrier per step (ping-pong buffers)
            const _Float16* hc = hping[t & 1];
            _Float16*       hn = hping[(t + 1) & 1];
            v16h aE0 = frag_a(eb, 64, lane, 0);
            v16h aE1 = frag_a(eb, 64, lane, 1);
            v16h aH0 = frag_a(hc, 64, lane, 0);
            v16h aH1 = frag_a(hc, 64, lane, 1);
            v8f g4[4];
#pragma unroll
            for (int g = 0; g < 4; ++g) {
                v8f acc = splat8(gbias[g]);
                acc = wmma16(aE0, bIh[g][0], acc);
                acc = wmma16(aE1, bIh[g][1], acc);
                acc = wmma16(aH0, bHh[g][0], acc);
                acc = wmma16(aH1, bHh[g][1], acc);
                g4[g] = acc;
            }
#pragma unroll
            for (int r = 0; r < 8; ++r) {
                float c_ = sigmoidf_(g4[1][r]) * creg[r] + sigmoidf_(g4[0][r]) * tanh_(g4[2][r]);
                creg[r] = c_;
                hn[(r + kb) * HID + col] = (_Float16)(sigmoidf_(g4[3][r]) * tanh_(c_));
            }
        }
    }
    __syncthreads();
    const _Float16* hT = hping[TSTEPS & 1];   // 30 even -> hping[0]

    // ------------------------- mean / logvar MLPs -------------------------
    for (int which = 0; which < 2; ++which) {
        const _Float16* Ws = which ? wLvS : wMeanS;
        const _Float16* Wf = which ? wLvF : wMeanF;
        const float* bs = which ? lv_bs : mean_bs;
        const float* bf = which ? lv_bf : mean_bf;
        float* dstL = which ? lv_s : mu_s;
        float* outL = which ? out_lv : out_mu;

        const _Float16* src = hT;
        for (int layer = 0; layer < 3; ++layer) {
            _Float16* dst = (layer & 1) ? m1 : m0;
            v16h a0 = frag_a(src, 64, lane, 0);
            v16h a1 = frag_a(src, 64, lane, 1);
            __syncthreads();
            int nt = layer * 4 + wave;
            int nl = wave * 16 + (lane & 15);
            v8f acc = splat8(bs[layer * 64 + nl]);
            acc = wmma16(a0, frag_b(Ws, 64, lane, nt, 0), acc);
            acc = wmma16(a1, frag_b(Ws, 64, lane, nt, 1), acc);
#pragma unroll
            for (int r = 0; r < 8; ++r) dst[(r + kb) * HID + nl] = (_Float16)leakyf_(acc[r]);
            src = dst;
            __syncthreads();
        }
        // final layer: H -> LATP (7 N-tiles over 4 waves)
        v16h a0 = frag_a(src, 64, lane, 0);
        v16h a1 = frag_a(src, 64, lane, 1);
#pragma unroll
        for (int part = 0; part < 2; ++part) {
            int nt = wave + part * 4;
            if (nt >= LATP / 16) continue;
            int nl = nt * 16 + (lane & 15);
            v8f acc = splat8(nl < LAT ? bf[nl] : 0.0f);
            acc = wmma16(a0, frag_b(Wf, 64, lane, nt, 0), acc);
            acc = wmma16(a1, frag_b(Wf, 64, lane, nt, 1), acc);
#pragma unroll
            for (int r = 0; r < 8; ++r) {
                float v = leakyf_(acc[r]);
                dstL[(r + kb) * LATP + nl] = v;
                if (nl < LAT) outL[(size_t)(b0 + r + kb) * LAT + nl] = v;
            }
        }
        __syncthreads();
    }

    // ----------------- z = tanh(eps*exp(0.5*lv)+mu); dh = z@W_init.T+b ----
    for (int u = tid; u < 16 * LATK; u += 128) {
        int r = u >> 7, l = u & 127;
        float v = 0.0f;
        if (l < LAT) {
            float e = eps[(size_t)(b0 + r) * LAT + l];
            v = tanh_(e * fexp_(0.5f * lv_s[r * LATP + l]) + mu_s[r * LATP + l]);
        }
        zbuf[u] = (_Float16)v;
    }
    if (tid < 64) prev_s[tid] = init_state[(size_t)(b0 + (tid >> 2)) * 4 + (tid & 3)];
    __syncthreads();
    {
        v16h az0 = frag_a(zbuf, LATK, lane, 0);
        v16h az1 = frag_a(zbuf, LATK, lane, 1);
        v16h az2 = frag_a(zbuf, LATK, lane, 2);
        v16h az3 = frag_a(zbuf, LATK, lane, 3);
        int nl = wave * 16 + (lane & 15);
        v8f acc = splat8(b_init[nl]);
        acc = wmma16(az0, frag_b(wInit, LATK, lane, wave, 0), acc);
        acc = wmma16(az1, frag_b(wInit, LATK, lane, wave, 1), acc);
        acc = wmma16(az2, frag_b(wInit, LATK, lane, wave, 2), acc);
        acc = wmma16(az3, frag_b(wInit, LATK, lane, wave, 3), acc);
#pragma unroll
        for (int r = 0; r < 8; ++r) {
            creg[r] = acc[r];                                    // c0 = dh
            hping[0][(r + kb) * HID + nl] = (_Float16)acc[r];    // h0 = dh
        }
    }
    __syncthreads();

    // ----------------------------- decoder -------------------------------
    {
        v16h bIh[4][2], bHh[4][2];
        float gbias[4];
#pragma unroll
        for (int g = 0; g < 4; ++g) {
            int nt = g * 4 + wave;
            int nl = nt * 16 + (lane & 15);
            gbias[g] = b_ih_d[nl] + b_hh_d[nl];
#pragma unroll
            for (int kt = 0; kt < 2; ++kt) {
                bIh[g][kt] = frag_b(wIhD, 64, lane, nt, kt);
                bHh[g][kt] = frag_b(wHhD, 64, lane, nt, kt);
            }
        }
        for (int t = 0; t < TSTEPS; ++t) {
            _Float16* eb = embp[t & 1];
            for (int u = tid; u < 16 * EMB; u += 128) {   // xe = prev @ W_se.T + b_se
                int r = u >> 6, c = u & 63;
                const float* wr = wse_s + c * 4;
                eb[u] = (_Float16)(bse_s[c] + prev_s[r * 4] * wr[0] + prev_s[r * 4 + 1] * wr[1]
                                 + prev_s[r * 4 + 2] * wr[2] + prev_s[r * 4 + 3] * wr[3]);
            }
            __syncthreads();
            const _Float16* hc = hping[t & 1];
            _Float16*       hn = hping[(t + 1) & 1];
            v16h aE0 = frag_a(eb, 64, lane, 0);
            v16h aE1 = frag_a(eb, 64, lane, 1);
            v16h aH0 = frag_a(hc, 64, lane, 0);
            v16h aH1 = frag_a(hc, 64, lane, 1);
            v8f g4[4];
#pragma unroll
            for (int g = 0; g < 4; ++g) {
                v8f acc = splat8(gbias[g]);
                acc = wmma16(aE0, bIh[g][0], acc);
                acc = wmma16(aE1, bIh[g][1], acc);
                acc = wmma16(aH0, bHh[g][0], acc);
                acc = wmma16(aH1, bHh[g][1], acc);
                g4[g] = acc;
            }
#pragma unroll
            for (int r = 0; r < 8; ++r) {
                float c_ = sigmoidf_(g4[1][r]) * creg[r] + sigmoidf_(g4[0][r]) * tanh_(g4[2][r]);
                creg[r] = c_;
                hn[(r + kb) * HID + col] = (_Float16)(sigmoidf_(g4[3][r]) * tanh_(c_));
            }
            __syncthreads();
            if (tid < 16) {   // controller head + bicycle-model integration
                int s = tid;
                float p0 = bc_s[0], p1 = bc_s[1];
                for (int k = 0; k < HID; ++k) {
                    float hv = (float)hn[s * HID + k];
                    p0 += hv * wc_s[k];
                    p1 += hv * wc_s[HID + k];
                }
                float steer = fminf(0.5f, fmaxf(-0.5f, p1));
                float x = prev_s[s * 4 + 0], y = prev_s[s * 4 + 1];
                float psi = prev_s[s * 4 + 2], v = prev_s[s * 4 + 3];
                float v1   = fminf(30.0f, fmaxf(0.0f, v + p0 * DTC));
                float psid = fminf(1.57f, fmaxf(-1.57f, v * __tanf(steer) * (1.0f / 2.5f)));
                float nx = x + v * __cosf(psi) * DTC;
                float ny = y + v * __sinf(psi) * DTC;
                float np = psi + psid * DTC;
                prev_s[s * 4 + 0] = nx; prev_s[s * 4 + 1] = ny;
                prev_s[s * 4 + 2] = np; prev_s[s * 4 + 3] = v1;
                float* o = out_rec + (size_t)(b0 + s) * (TSTEPS * 4) + t * 4;
                o[0] = nx; o[1] = ny; o[2] = np; o[3] = v1;
            }
            __syncthreads();
        }
    }
}

extern "C" void kernel_launch(void* const* d_in, const int* in_sizes, int n_in,
                              void* d_out, int out_size, void* d_ws, size_t ws_size,
                              hipStream_t stream) {
    (void)n_in; (void)out_size; (void)ws_size;
    const int B = in_sizes[0] / (TSTEPS * 4);

    // f16 weight regions in workspace (all offsets 16B aligned).
    _Float16* w = (_Float16*)d_ws;
    size_t o = 0;
    _Float16* wIhE   = w + o; o += 256 * 64;
    _Float16* wHhE   = w + o; o += 256 * 64;
    _Float16* wMeanS = w + o; o += 192 * 64;
    _Float16* wMeanF = w + o; o += LATP * 64;
    _Float16* wLvS   = w + o; o += 192 * 64;
    _Float16* wLvF   = w + o; o += LATP * 64;
    _Float16* wInit  = w + o; o += 64 * LATK;
    _Float16* wIhD   = w + o; o += 256 * 64;
    _Float16* wHhD   = w + o; o += 256 * 64;   // total ~220 KB

    auto cvt = [&](const void* src, _Float16* dst, int N, int K, int Np, int Kp) {
        int total = Np * Kp;
        cvt_pad_kernel<<<(total + 255) / 256, 256, 0, stream>>>((const float*)src, dst, N, K, Np, Kp);
    };
    cvt(d_in[5],  wIhE,   256, 64, 256, 64);
    cvt(d_in[6],  wHhE,   256, 64, 256, 64);
    cvt(d_in[9],  wMeanS, 192, 64, 192, 64);
    cvt(d_in[11], wMeanF, LAT, 64, LATP, 64);
    cvt(d_in[13], wLvS,   192, 64, 192, 64);
    cvt(d_in[15], wLvF,   LAT, 64, LATP, 64);
    cvt(d_in[17], wInit,  64, LAT, 64, LATK);
    cvt(d_in[19], wIhD,   256, 64, 256, 64);
    cvt(d_in[20], wHhD,   256, 64, 256, 64);

    vae_lstm_kernel<<<B / 16, 128, 0, stream>>>(
        (const float*)d_in[0], (const float*)d_in[1], (const float*)d_in[2],
        (const float*)d_in[3], (const float*)d_in[4],
        (const float*)d_in[7], (const float*)d_in[8],
        (const float*)d_in[10], (const float*)d_in[12],
        (const float*)d_in[14], (const float*)d_in[16],
        (const float*)d_in[18],
        (const float*)d_in[21], (const float*)d_in[22],
        (const float*)d_in[23], (const float*)d_in[24],
        wIhE, wHhE, wMeanS, wMeanF, wLvS, wLvF, wInit, wIhD, wHhD,
        (float*)d_out, B);
}
